// BidirectionalMambaBlock_17583596109965
// MI455X (gfx1250) — matrix operand order repaired and checked by reading, not compiled, not run on hardware
//
#include <hip/hip_runtime.h>
#include <hip/hip_bf16.h>
#include <cstdint>

// ---------------------------------------------------------------------------
// Bidirectional Mamba block for gfx1250 (MI455X).
// WMMA f16->f32 GEMMs with double-buffered async global->LDS tile staging
// (global_load_async_to_lds_b128 / s_wait_asynccnt).
// ---------------------------------------------------------------------------

typedef __attribute__((ext_vector_type(16))) _Float16 v16h;
typedef __attribute__((ext_vector_type(8)))  float    v8f;

// Problem constants (match reference)
constexpr int B_      = 4;
constexpr int L_      = 2048;
constexpr int DM      = 512;    // d_model
constexpr int DI      = 1024;   // d_inner
constexpr int DSTATE  = 16;
constexpr int DCONV   = 4;
constexpr int DTRANK  = 32;
constexpr int DBLC    = DTRANK + 2 * DSTATE; // 64
constexpr int MROWS   = B_ * L_;             // 8192

// ---------------- workspace layout (all sizes multiples of 256B) ----------
constexpr size_t S_XN   = (size_t)MROWS * DM * 2;      // xn f16
constexpr size_t S_XZ   = (size_t)MROWS * 2 * DI * 2;  // xz f16 [M, 2048]
constexpr size_t S_XC   = (size_t)MROWS * DI * 2;      // xc f16
constexpr size_t S_DBL  = (size_t)MROWS * DBLC * 4;    // dbl f32
constexpr size_t S_DTR  = (size_t)MROWS * DTRANK * 2;  // dtr f16
constexpr size_t S_DT   = (size_t)MROWS * DI * 4;      // dt f32
constexpr size_t S_Y    = (size_t)MROWS * DI * 2;      // y f16
constexpr size_t S_DIR  = (size_t)MROWS * DM * 2;      // f/b f16
constexpr size_t S_CAT  = (size_t)MROWS * DI * 2;      // concat f16
constexpr size_t S_WIN  = (size_t)(2 * DI) * DM * 2;   // in_w f16
constexpr size_t S_WXP  = (size_t)DBLC * DI * 2;       // xproj f16
constexpr size_t S_WDT  = (size_t)DI * DTRANK * 2;     // dt_w f16
constexpr size_t S_WOUT = (size_t)DM * DI * 2;         // out_w f16
constexpr size_t S_WMG  = (size_t)DM * (2 * DM) * 2;   // merge_w f16

constexpr size_t O_XN   = 0;
constexpr size_t O_XNR  = O_XN  + S_XN;
constexpr size_t O_XZ   = O_XNR + S_XN;
constexpr size_t O_XC   = O_XZ  + S_XZ;
constexpr size_t O_DBL  = O_XC  + S_XC;
constexpr size_t O_DTR  = O_DBL + S_DBL;
constexpr size_t O_DT   = O_DTR + S_DTR;
constexpr size_t O_Y    = O_DT  + S_DT;
constexpr size_t O_F    = O_Y   + S_Y;
constexpr size_t O_Bf   = O_F   + S_DIR;
constexpr size_t O_CAT  = O_Bf  + S_DIR;
constexpr size_t O_WIN  = O_CAT + S_CAT;
constexpr size_t O_WXP  = O_WIN + S_WIN;
constexpr size_t O_WDT  = O_WXP + S_WXP;
constexpr size_t O_WOUT = O_WDT + S_WDT;
constexpr size_t O_WMG  = O_WOUT + S_WOUT;

__device__ __forceinline__ float sigmoidf_(float v) {
    return 1.0f / (1.0f + __expf(-v));
}

// Async global -> LDS 16B copy (per lane), tracked by ASYNCcnt.
__device__ __forceinline__ void async_copy_b128(const _Float16* gptr,
                                                _Float16* lptr) {
    const uint32_t laddr = (uint32_t)(uintptr_t)lptr;   // low 32 = LDS offset
    const uint64_t gaddr = (uint64_t)(uintptr_t)gptr;
    asm volatile("global_load_async_to_lds_b128 %0, %1, off"
                 :
                 : "v"(laddr), "v"(gaddr)
                 : "memory");
}

__device__ __forceinline__ void wait_asynccnt0() {
    asm volatile("s_wait_asynccnt 0x0" ::: "memory");
}

// ---------------------------------------------------------------------------
// f32 -> f16 conversion (grid-stride)
// ---------------------------------------------------------------------------
__global__ void cvt_f32_f16_kernel(const float* __restrict__ src,
                                   _Float16* __restrict__ dst, int n) {
    for (int i = blockIdx.x * blockDim.x + threadIdx.x; i < n;
         i += gridDim.x * blockDim.x) {
        dst[i] = (_Float16)src[i];
    }
}

// ---------------------------------------------------------------------------
// LayerNorm over D_MODEL, emits f16 normalized rows + time-flipped copy.
// One block (256 thr) per row of 512.
// ---------------------------------------------------------------------------
__global__ __launch_bounds__(256) void layernorm_kernel(
    const float* __restrict__ x, const float* __restrict__ g,
    const float* __restrict__ bb, _Float16* __restrict__ xn,
    _Float16* __restrict__ xnr) {
    const int row = blockIdx.x;
    const int tid = threadIdx.x;
    const float* xr = x + (size_t)row * DM;
    float v0 = xr[tid];
    float v1 = xr[tid + 256];

    __shared__ float s1[256];
    __shared__ float s2[256];
    s1[tid] = v0 + v1;
    s2[tid] = v0 * v0 + v1 * v1;
    __syncthreads();
    for (int off = 128; off > 0; off >>= 1) {
        if (tid < off) { s1[tid] += s1[tid + off]; s2[tid] += s2[tid + off]; }
        __syncthreads();
    }
    const float mu  = s1[0] * (1.0f / DM);
    const float var = s2[0] * (1.0f / DM) - mu * mu;
    const float rs  = rsqrtf(var + 1e-5f);

    const int b = row >> 11;          // row / L
    const int t = row & (L_ - 1);     // row % L
    const size_t rrow = ((size_t)b << 11) + (size_t)(L_ - 1 - t);

    _Float16 o0 = (_Float16)((v0 - mu) * rs * g[tid]       + bb[tid]);
    _Float16 o1 = (_Float16)((v1 - mu) * rs * g[tid + 256] + bb[tid + 256]);
    xn [(size_t)row * DM + tid]        = o0;
    xn [(size_t)row * DM + tid + 256]  = o1;
    xnr[rrow * DM + tid]               = o0;
    xnr[rrow * DM + tid + 256]         = o1;
}

// ---------------------------------------------------------------------------
// Tiled WMMA GEMM: C[M,N] = A[M,K] * W[N,K]^T  (f16 in, f32 accumulate)
// Block tile 128x128, BK=32; 8 waves (wave32) in a 2(M)x4(N) grid; each wave
// owns a 64x32 sub-tile = 4x2 fragments of 16x16. Tile staging uses
// double-buffered global_load_async_to_lds_b128 (ASYNCcnt). Optional bias,
// f32 residual, output f16 or f32.
// ---------------------------------------------------------------------------
union Frag16 { v16h h; uint4 q[2]; };

constexpr int GBM = 128, GBN = 128, GBK = 32, LDT = 40; // LDS stride (halfs)

__device__ __forceinline__ void stage_tile_async(
    const _Float16* __restrict__ A, const _Float16* __restrict__ W,
    _Float16* sA, _Float16* sB, int tileM, int tileN, int kb, int N, int K,
    int tid) {
#pragma unroll
    for (int i = 0; i < 2; ++i) {
        const int c   = tid + i * 256;   // 0..511 chunk id
        const int row = c >> 2;          // 0..127
        const int kc  = (c & 3) * 8;     // 0..24
        const int kk  = kb + kc;
        _Float16* dA = &sA[row * LDT + kc];
        if (kk < K) {
            async_copy_b128(A + (size_t)(tileM + row) * K + kk, dA);
        } else {
            *reinterpret_cast<uint4*>(dA) = (uint4){0, 0, 0, 0};
        }
        _Float16* dB = &sB[row * LDT + kc];
        const int n = tileN + row;
        if (n < N && kk < K) {
            async_copy_b128(W + (size_t)n * K + kk, dB);
        } else {
            *reinterpret_cast<uint4*>(dB) = (uint4){0, 0, 0, 0};
        }
    }
}

template <bool OUT_F16>
__global__ __launch_bounds__(256) void gemm_wmma_kernel(
    const _Float16* __restrict__ A, const _Float16* __restrict__ W,
    void* __restrict__ Cout, int M, int N, int K,
    const float* __restrict__ bias, const float* __restrict__ residual) {
    __shared__ __align__(16) _Float16 shA[2][GBM * LDT]; // 2 x 10 KB
    __shared__ __align__(16) _Float16 shB[2][GBN * LDT]; // 2 x 10 KB

    const int tid   = threadIdx.x;
    const int tileM = blockIdx.x * GBM;
    const int tileN = blockIdx.y * GBN;
    const int lane  = tid & 31;
    const int wave  = tid >> 5;
    const int wm    = wave & 1;   // 0..1 over M (64 rows each)
    const int wn    = wave >> 1;  // 0..3 over N (32 cols each)
    const int lr    = lane & 15;
    const int lh    = lane >> 4;

    v8f acc[4][2];
#pragma unroll
    for (int i = 0; i < 4; ++i)
#pragma unroll
        for (int j = 0; j < 2; ++j)
            acc[i][j] = (v8f){0.f, 0.f, 0.f, 0.f, 0.f, 0.f, 0.f, 0.f};

    // Prologue: stage first k-tile into buffer 0.
    stage_tile_async(A, W, shA[0], shB[0], tileM, tileN, 0, N, K, tid);
    wait_asynccnt0();
    __syncthreads();

    int cur = 0;
    for (int kb = 0; kb < K; kb += GBK, cur ^= 1) {
        // Issue async staging of the next k-tile into the other buffer.
        if (kb + GBK < K)
            stage_tile_async(A, W, shA[cur ^ 1], shB[cur ^ 1], tileM, tileN,
                             kb + GBK, N, K, tid);

        // Compute on the current buffer.
        Frag16 bf[2];
#pragma unroll
        for (int fn = 0; fn < 2; ++fn) {
            const int base = (wn * 32 + fn * 16 + lr) * LDT + lh * 16;
            bf[fn].q[0] = *reinterpret_cast<const uint4*>(&shB[cur][base]);
            bf[fn].q[1] = *reinterpret_cast<const uint4*>(&shB[cur][base + 8]);
        }
#pragma unroll
        for (int fm = 0; fm < 4; ++fm) {
            Frag16 af;
            const int base = (wm * 64 + fm * 16 + lr) * LDT + lh * 16;
            af.q[0] = *reinterpret_cast<const uint4*>(&shA[cur][base]);
            af.q[1] = *reinterpret_cast<const uint4*>(&shA[cur][base + 8]);
#pragma unroll
            for (int fn = 0; fn < 2; ++fn) {
                acc[fm][fn] = __builtin_amdgcn_wmma_f32_16x16x32_f16(
                    false, af.h, false, bf[fn].h, (short)0, acc[fm][fn], false,
                    false);
            }
        }

        // Order: this tile's reads precede next overwrite; next tile's async
        // writes complete before anyone reads them.
        wait_asynccnt0();
        __syncthreads();
    }

    // epilogue: C/D layout -> lane lr = col, row = lh*8 + v
#pragma unroll
    for (int fm = 0; fm < 4; ++fm) {
#pragma unroll
        for (int fn = 0; fn < 2; ++fn) {
            const int gn = tileN + wn * 32 + fn * 16 + lr;
            if (gn >= N) continue;
            const float bv = bias ? bias[gn] : 0.0f;
#pragma unroll
            for (int v = 0; v < 8; ++v) {
                const int gm = tileM + wm * 64 + fm * 16 + lh * 8 + v;
                float val = acc[fm][fn][v] + bv;
                if (residual) val += residual[(size_t)gm * N + gn];
                if constexpr (OUT_F16)
                    reinterpret_cast<_Float16*>(Cout)[(size_t)gm * N + gn] =
                        (_Float16)val;
                else
                    reinterpret_cast<float*>(Cout)[(size_t)gm * N + gn] = val;
            }
        }
    }
}

// ---------------------------------------------------------------------------
// Depthwise causal conv (width 4) + SiLU. xi lives in xz cols [0,DI),
// row stride 2*DI. One thread per (m, d).
// ---------------------------------------------------------------------------
__global__ __launch_bounds__(256) void conv_silu_kernel(
    const _Float16* __restrict__ xz, const float* __restrict__ cw,
    const float* __restrict__ cb, _Float16* __restrict__ xc) {
    const int idx = blockIdx.x * blockDim.x + threadIdx.x;
    const int d = idx & (DI - 1);
    const int m = idx >> 10;
    const int t = m & (L_ - 1);
    float acc = cb[d];
#pragma unroll
    for (int k = 0; k < DCONV; ++k) {
        const int dt_ = k - (DCONV - 1);
        if (t + dt_ >= 0)
            acc += (float)xz[(size_t)(m + dt_) * (2 * DI) + d] * cw[d * DCONV + k];
    }
    xc[(size_t)m * DI + d] = (_Float16)(acc * sigmoidf_(acc));
}

// dbl[:, :32] f32 -> dtr f16
__global__ void slice_dtr_kernel(const float* __restrict__ dbl,
                                 _Float16* __restrict__ dtr) {
    const int idx = blockIdx.x * blockDim.x + threadIdx.x;
    const int m = idx >> 5;
    const int r = idx & 31;
    dtr[idx] = (_Float16)dbl[(size_t)m * DBLC + r];
}

// in-place softplus (bias already fused into GEMM epilogue)
__global__ void softplus_kernel(float* __restrict__ dt, int n) {
    const int idx = blockIdx.x * blockDim.x + threadIdx.x;
    if (idx < n) {
        const float v = dt[idx];
        dt[idx] = (v > 20.0f) ? v : log1pf(__expf(v));
    }
}

// ---------------------------------------------------------------------------
// Selective scan: one thread per (b, d) channel, sequential over L.
// B/C vectors (dbl cols 32..63) broadcast per timestep through LDS.
// Fuses y = (y_ssm + xc*Dp) * silu(z); emits f16.
// ---------------------------------------------------------------------------
__global__ __launch_bounds__(256) void scan_kernel(
    const float* __restrict__ dbl, const float* __restrict__ dt,
    const _Float16* __restrict__ xc, const _Float16* __restrict__ xz,
    const float* __restrict__ A_log, const float* __restrict__ Dp,
    _Float16* __restrict__ y) {
    const int b = blockIdx.y;
    const int d = blockIdx.x * blockDim.x + threadIdx.x;

    float Arow[DSTATE];
#pragma unroll
    for (int s = 0; s < DSTATE; ++s)
        Arow[s] = -__expf(A_log[(size_t)d * DSTATE + s]);
    float h[DSTATE];
#pragma unroll
    for (int s = 0; s < DSTATE; ++s) h[s] = 0.0f;
    const float dpd = Dp[d];

    __shared__ float shBC[2 * DSTATE];
    for (int t = 0; t < L_; ++t) {
        const size_t m = (size_t)b * L_ + t;
        if (threadIdx.x < 2 * DSTATE)
            shBC[threadIdx.x] = dbl[m * DBLC + DTRANK + threadIdx.x];
        __syncthreads();

        const float dtv = dt[m * DI + d];
        const float xv  = (float)xc[m * DI + d];
        const float dtx = dtv * xv;
        float yv = 0.0f;
#pragma unroll
        for (int s = 0; s < DSTATE; ++s) {
            const float dA = __expf(dtv * Arow[s]);
            h[s] = dA * h[s] + dtx * shBC[s];
            yv += h[s] * shBC[DSTATE + s];
        }
        yv += xv * dpd;
        const float zv = (float)xz[m * (2 * DI) + DI + d];
        yv *= zv * sigmoidf_(zv);
        y[m * DI + d] = (_Float16)yv;
        __syncthreads();
    }
}

// concat [f | flip(b)] -> cat f16 [M, 2*DM]
__global__ void cat_kernel(const _Float16* __restrict__ f,
                           const _Float16* __restrict__ bb,
                           _Float16* __restrict__ cat) {
    const int idx = blockIdx.x * blockDim.x + threadIdx.x;
    const int c = idx & (2 * DM - 1);
    const int m = idx >> 10;
    const int b = m >> 11;
    const int t = m & (L_ - 1);
    _Float16 v;
    if (c < DM) {
        v = f[(size_t)m * DM + c];
    } else {
        const size_t rm = ((size_t)b << 11) + (size_t)(L_ - 1 - t);
        v = bb[rm * DM + (c - DM)];
    }
    cat[idx] = v;
}

// ---------------------------------------------------------------------------
// Host side
// ---------------------------------------------------------------------------
static inline void launch_gemm(bool outF16, const _Float16* A, const _Float16* W,
                               void* C, int M, int N, int K, const float* bias,
                               const float* resid, hipStream_t s) {
    dim3 grid(M / 128, (N + 127) / 128), blk(256);
    if (outF16)
        gemm_wmma_kernel<true><<<grid, blk, 0, s>>>(A, W, C, M, N, K, bias, resid);
    else
        gemm_wmma_kernel<false><<<grid, blk, 0, s>>>(A, W, C, M, N, K, bias, resid);
}

static inline void cvt(const float* src, _Float16* dst, int n, hipStream_t s) {
    int blocks = (n + 255) / 256;
    if (blocks > 4096) blocks = 4096;
    cvt_f32_f16_kernel<<<blocks, 256, 0, s>>>(src, dst, n);
}

extern "C" void kernel_launch(void* const* d_in, const int* in_sizes, int n_in,
                              void* d_out, int out_size, void* d_ws,
                              size_t ws_size, hipStream_t stream) {
    (void)in_sizes; (void)n_in; (void)out_size; (void)ws_size;

    const float* x       = (const float*)d_in[0];
    const float* norm_g  = (const float*)d_in[1];
    const float* norm_b  = (const float*)d_in[2];
    const float* merge_w = (const float*)d_in[21];
    const float* merge_b = (const float*)d_in[22];

    char* ws = (char*)d_ws;
    _Float16* xn16   = (_Float16*)(ws + O_XN);
    _Float16* xnr16  = (_Float16*)(ws + O_XNR);
    _Float16* xz16   = (_Float16*)(ws + O_XZ);
    _Float16* xc16   = (_Float16*)(ws + O_XC);
    float*    dblF   = (float*)   (ws + O_DBL);
    _Float16* dtr16  = (_Float16*)(ws + O_DTR);
    float*    dtF    = (float*)   (ws + O_DT);
    _Float16* y16    = (_Float16*)(ws + O_Y);
    _Float16* f16b   = (_Float16*)(ws + O_F);
    _Float16* b16b   = (_Float16*)(ws + O_Bf);
    _Float16* cat16  = (_Float16*)(ws + O_CAT);
    _Float16* wIn16  = (_Float16*)(ws + O_WIN);
    _Float16* wXp16  = (_Float16*)(ws + O_WXP);
    _Float16* wDt16  = (_Float16*)(ws + O_WDT);
    _Float16* wOut16 = (_Float16*)(ws + O_WOUT);
    _Float16* wMg16  = (_Float16*)(ws + O_WMG);

    // LayerNorm -> xn (f16) and time-flipped copy
    layernorm_kernel<<<MROWS, 256, 0, stream>>>(x, norm_g, norm_b, xn16, xnr16);
    // merge weight once
    cvt(merge_w, wMg16, DM * 2 * DM, stream);

    for (int dir = 0; dir < 2; ++dir) {
        const int pb = 3 + dir * 9; // param base index
        const float* in_w    = (const float*)d_in[pb + 0];
        const float* conv_w  = (const float*)d_in[pb + 1];
        const float* conv_b  = (const float*)d_in[pb + 2];
        const float* xproj_w = (const float*)d_in[pb + 3];
        const float* dt_w    = (const float*)d_in[pb + 4];
        const float* dt_b    = (const float*)d_in[pb + 5];
        const float* A_log   = (const float*)d_in[pb + 6];
        const float* Dp      = (const float*)d_in[pb + 7];
        const float* out_w   = (const float*)d_in[pb + 8];

        cvt(in_w,    wIn16,  2 * DI * DM,  stream);
        cvt(xproj_w, wXp16,  DBLC * DI,    stream);
        cvt(dt_w,    wDt16,  DI * DTRANK,  stream);
        cvt(out_w,   wOut16, DM * DI,      stream);

        const _Float16* src = dir ? xnr16 : xn16;

        // in_proj: [M,512] x [2048,512]^T -> xz f16 [M,2048]
        launch_gemm(true, src, wIn16, xz16, MROWS, 2 * DI, DM, nullptr, nullptr,
                    stream);
        // depthwise conv + SiLU -> xc f16
        conv_silu_kernel<<<(MROWS * DI) / 256, 256, 0, stream>>>(xz16, conv_w,
                                                                 conv_b, xc16);
        // x_proj: [M,1024] x [64,1024]^T -> dbl f32 [M,64]
        launch_gemm(false, xc16, wXp16, dblF, MROWS, DBLC, DI, nullptr, nullptr,
                    stream);
        // dtr slice -> f16
        slice_dtr_kernel<<<(MROWS * DTRANK) / 256, 256, 0, stream>>>(dblF, dtr16);
        // dt proj (+dt_b fused): [M,32] x [1024,32]^T -> dt f32 [M,1024]
        launch_gemm(false, dtr16, wDt16, dtF, MROWS, DI, DTRANK, dt_b, nullptr,
                    stream);
        softplus_kernel<<<(MROWS * DI) / 256, 256, 0, stream>>>(dtF, MROWS * DI);
        // selective scan (fused Dp residual + silu(z) gating) -> y f16
        scan_kernel<<<dim3(DI / 256, B_), 256, 0, stream>>>(
            dblF, dtF, xc16, xz16, A_log, Dp, y16);
        // out_proj: [M,1024] x [512,1024]^T -> dir output f16 [M,512]
        launch_gemm(true, y16, wOut16, dir ? (void*)b16b : (void*)f16b, MROWS,
                    DM, DI, nullptr, nullptr, stream);
    }

    // concat [f | flip(b)] and merge GEMM with fused residual x + merge_b
    cat_kernel<<<(MROWS * 2 * DM) / 256, 256, 0, stream>>>(f16b, b16b, cat16);
    launch_gemm(false, cat16, wMg16, d_out, MROWS, DM, 2 * DM, merge_b, x,
                stream);
}
